// Self_Attn_5291399708918
// MI455X (gfx1250) — matrix operand order repaired and checked by reading, not compiled, not run on hardware
//
#include <hip/hip_runtime.h>
#include <hip/hip_bf16.h>

// Problem constants (match reference)
#define BDIM 8
#define CDIM 512
#define NDIM 2048

typedef __attribute__((ext_vector_type(16))) __bf16 v16bf;
typedef __attribute__((ext_vector_type(8)))  __bf16 v8bf;
typedef __attribute__((ext_vector_type(4)))  __bf16 v4bf;
typedef __attribute__((ext_vector_type(8)))  float  v8f;

// ---------- bf16 helpers (bit-exact RNE, no reliance on cvt intrinsics) ----------
__device__ inline __bf16 f2bf(float f) {
    unsigned u = __builtin_bit_cast(unsigned, f);
    unsigned r = u + 0x7FFFu + ((u >> 16) & 1u);
    unsigned short h = (unsigned short)(r >> 16);
    return __builtin_bit_cast(__bf16, h);
}
__device__ inline float bf2f(__bf16 b) {
    unsigned short h = __builtin_bit_cast(unsigned short, b);
    unsigned u = ((unsigned)h) << 16;
    return __builtin_bit_cast(float, u);
}

// ---------- CDNA5 async global->LDS copy (16 bytes per issue), ASYNCcnt-tracked ----------
__device__ inline void async_copy_b128(unsigned lds_off, const void* gaddr) {
    asm volatile("global_load_async_to_lds_b128 %0, %1, off"
                 :: "v"(lds_off), "v"(gaddr) : "memory");
}
__device__ inline void wait_asynccnt0() {
    asm volatile("s_wait_asynccnt 0x0" ::: "memory");
}
// Low 32 bits of a generic LDS pointer == byte offset within the workgroup LDS allocation.
__device__ inline unsigned lds_offset(const void* p) {
    return (unsigned)(size_t)p;
}

// ---------- WMMA fragment loaders (contiguous k = 16*(lane/16) + s, same order for A and B) ----------
// A: 16x32 (MxK). Lanes 0-15: row M=lane, halves 0..15 = K0..15.
//    Lanes 16-31: row M=lane-16, halves 0..15 = K16..31.
// Works for both global and LDS (generic) pointers.
__device__ inline v16bf load_frag_a(const __bf16* base, int ld, int lane) {
    int m    = lane & 15;
    int hsel = lane >> 4;                 // 0 or 1
    const __bf16* p = base + (size_t)m * ld + hsel * 16;
    v8bf lo = *(const v8bf*)(p);
    v8bf hi = *(const v8bf*)(p + 8);
    v16bf a;
#pragma unroll
    for (int i = 0; i < 8; ++i) { a[i] = lo[i]; a[i + 8] = hi[i]; }
    return a;
}
// B: 32x16 (KxN). Lane holds one column (N=lane&15): lanes 0-15 K=0..15, lanes 16-31 K=16..31.
// We feed it rows of the K-contiguous "B-transposed" source: source row i == B column i.
__device__ inline v16bf load_frag_b(const __bf16* __restrict__ base, int ld, int lane) {
    int n    = lane & 15;
    int hsel = lane >> 4;                 // 0 or 1
    const __bf16* p = base + (size_t)n * ld + hsel * 16;
    return *(const v16bf*)p;
}
__device__ inline v8f wmma_bf16(v16bf a, v16bf b, v8f c) {
    return __builtin_amdgcn_wmma_f32_16x16x32_bf16(
        /*neg_a=*/false, a, /*neg_b=*/false, b,
        /*c_mod=*/(short)0, c, /*reuse_a=*/false, /*reuse_b=*/false);
}

// ---------- Kernel 1: fp32 -> bf16 convert ----------
__global__ __launch_bounds__(256) void cvt_kernel(const float* __restrict__ s,
                                                  __bf16* __restrict__ d, int n) {
    int i = (blockIdx.x * 256 + threadIdx.x) * 4;
    if (i + 3 < n) {
        float4 f = *(const float4*)(s + i);
        v4bf o;
        o[0] = f2bf(f.x); o[1] = f2bf(f.y); o[2] = f2bf(f.z); o[3] = f2bf(f.w);
        *(v4bf*)(d + i) = o;
    } else {
        for (int j = i; j < n; ++j) d[j] = f2bf(s[j]);
    }
}

// ---------- Kernel 2: projection  P[b,o,c] = sum_n W[o,n]*X[b,c,n] + bias[o] ----------
// Block = (b, o-tile). The 16x2048 W strip (64KB, contiguous) is async-staged into LDS
// once and shared by all 8 waves; wave w computes the 16(o) x 64(c) strip at c0 = w*64.
// transposed==0: store bf16 P[b][o][c] (ld C).  transposed==1: store bf16 Pt[b][c][o] (ld N).
__global__ __launch_bounds__(256) void proj_kernel(const __bf16* __restrict__ Wh,
                                                   const __bf16* __restrict__ Xh,
                                                   const float*  __restrict__ bias,
                                                   __bf16* __restrict__ dst,
                                                   int transposed) {
    __shared__ __bf16 Atile[16 * NDIM];       // 64 KB
    const int lane = threadIdx.x & 31;
    const int wave = threadIdx.x >> 5;
    int ot = blockIdx.x & 127;                // N/16
    int b  = blockIdx.x >> 7;
    int o0 = ot * 16, c0 = wave * 64;

    // Stage A strip: 16 rows x 2048 cols bf16 == contiguous 64KB at Wh + o0*N.
    {
        const char* Ag = (const char*)(Wh + (size_t)o0 * NDIM);
        unsigned l0 = lds_offset(Atile);
#pragma unroll
        for (int j = 0; j < 16; ++j) {
            int ch = threadIdx.x + j * 256;   // 0..4095 chunks of 16B
            async_copy_b128(l0 + ch * 16, Ag + (size_t)ch * 16);
        }
        wait_asynccnt0();
    }
    __syncthreads();

    const __bf16* Bbase = Xh + (size_t)b * CDIM * NDIM + (size_t)c0 * NDIM;

    v8f acc0 = {}, acc1 = {}, acc2 = {}, acc3 = {};
    for (int k = 0; k < NDIM; k += 32) {
        __builtin_prefetch(Bbase + k + 256, 0, 3);
        v16bf a  = load_frag_a(Atile + k, NDIM, lane);          // from LDS
        v16bf b0 = load_frag_b(Bbase + k,                      NDIM, lane);
        v16bf b1 = load_frag_b(Bbase + k + (size_t)16 * NDIM,  NDIM, lane);
        v16bf b2 = load_frag_b(Bbase + k + (size_t)32 * NDIM,  NDIM, lane);
        v16bf b3 = load_frag_b(Bbase + k + (size_t)48 * NDIM,  NDIM, lane);
        acc0 = wmma_bf16(a, b0, acc0);
        acc1 = wmma_bf16(a, b1, acc1);
        acc2 = wmma_bf16(a, b2, acc2);
        acc3 = wmma_bf16(a, b3, acc3);
    }

    int nn = lane & 15;
    int mb = (lane >> 4) * 8;
    float bv[8];
#pragma unroll
    for (int r = 0; r < 8; ++r) bv[r] = bias[o0 + mb + r];

    v8f accs[4] = {acc0, acc1, acc2, acc3};
    if (!transposed) {
#pragma unroll
        for (int t = 0; t < 4; ++t) {
            __bf16* out = dst + (size_t)b * NDIM * CDIM + (size_t)o0 * CDIM + (c0 + t * 16 + nn);
#pragma unroll
            for (int r = 0; r < 8; ++r)
                out[(size_t)(mb + r) * CDIM] = f2bf(accs[t][r] + bv[r]);
        }
    } else {
#pragma unroll
        for (int t = 0; t < 4; ++t) {
            __bf16* out = dst + (size_t)b * CDIM * NDIM + (size_t)(c0 + t * 16 + nn) * NDIM + (o0 + mb);
#pragma unroll
            for (int r = 0; r < 8; ++r) out[r] = f2bf(accs[t][r] + bv[r]);
        }
    }
}

// ---------- Kernel 3: energy + softmax -> attn (bf16, row-major [b][n][m]) ----------
// One block per (b, 16 query rows). The 16x512 pq strip (16KB, contiguous) is async-staged
// into LDS and shared by all 8 waves; each wave covers 256 m-columns, 4 m-tiles in flight.
__global__ __launch_bounds__(256) void energy_softmax_kernel(const __bf16* __restrict__ pq,
                                                             const __bf16* __restrict__ pk,
                                                             __bf16* __restrict__ attn) {
    __shared__ __bf16 Aq[16 * CDIM];     // 16 KB
    __shared__ __bf16 S[16 * NDIM];      // 64 KB
    __shared__ float  red[16 * 16];
    __shared__ float  rowmax[16];
    __shared__ float  rowinv[16];

    const int lane = threadIdx.x & 31;
    const int wave = threadIdx.x >> 5;
    int nt = blockIdx.x & 127;
    int b  = blockIdx.x >> 7;
    int n0 = nt * 16;

    // Stage pq strip: 16 rows x 512 cols bf16 == contiguous 16KB.
    {
        const char* Ag = (const char*)(pq + (size_t)b * NDIM * CDIM + (size_t)n0 * CDIM);
        unsigned l0 = lds_offset(Aq);
#pragma unroll
        for (int j = 0; j < 4; ++j) {
            int ch = threadIdx.x + j * 256;   // 0..1023 chunks of 16B
            async_copy_b128(l0 + ch * 16, Ag + (size_t)ch * 16);
        }
        wait_asynccnt0();
    }
    __syncthreads();

    int nn = lane & 15;
    int mb = (lane >> 4) * 8;

    for (int mg = 0; mg < 4; ++mg) {                 // 4 groups of 64 m-columns
        int m0 = wave * 256 + mg * 64;
        const __bf16* Bbase = pk + (size_t)b * NDIM * CDIM + (size_t)m0 * CDIM;  // ld C
        v8f acc0 = {}, acc1 = {}, acc2 = {}, acc3 = {};
        for (int k = 0; k < CDIM; k += 32) {
            v16bf a  = load_frag_a(Aq + k, CDIM, lane);          // from LDS
            v16bf b0 = load_frag_b(Bbase + k,                     CDIM, lane);
            v16bf b1 = load_frag_b(Bbase + k + (size_t)16 * CDIM, CDIM, lane);
            v16bf b2 = load_frag_b(Bbase + k + (size_t)32 * CDIM, CDIM, lane);
            v16bf b3 = load_frag_b(Bbase + k + (size_t)48 * CDIM, CDIM, lane);
            acc0 = wmma_bf16(a, b0, acc0);
            acc1 = wmma_bf16(a, b1, acc1);
            acc2 = wmma_bf16(a, b2, acc2);
            acc3 = wmma_bf16(a, b3, acc3);
        }
        v8f accs[4] = {acc0, acc1, acc2, acc3};
#pragma unroll
        for (int t = 0; t < 4; ++t)
#pragma unroll
            for (int r = 0; r < 8; ++r)
                S[(mb + r) * NDIM + (m0 + t * 16 + nn)] = f2bf(accs[t][r]);
    }
    __syncthreads();

    // Row-wise softmax over m = 0..2047 (16 threads per row).
    int row = threadIdx.x >> 4;
    int sub = threadIdx.x & 15;
    float mx = -3.4e38f;
    for (int m = sub; m < NDIM; m += 16) mx = fmaxf(mx, bf2f(S[row * NDIM + m]));
    red[row * 16 + sub] = mx;
    __syncthreads();
    if (threadIdx.x < 16) {
        float v = red[threadIdx.x * 16];
#pragma unroll
        for (int i = 1; i < 16; ++i) v = fmaxf(v, red[threadIdx.x * 16 + i]);
        rowmax[threadIdx.x] = v;
    }
    __syncthreads();
    float rm = rowmax[row];
    float s = 0.f;
    for (int m = sub; m < NDIM; m += 16) s += __expf(bf2f(S[row * NDIM + m]) - rm);
    red[row * 16 + sub] = s;
    __syncthreads();
    if (threadIdx.x < 16) {
        float v = 0.f;
#pragma unroll
        for (int i = 0; i < 16; ++i) v += red[threadIdx.x * 16 + i];
        rowinv[threadIdx.x] = 1.0f / v;
    }
    __syncthreads();
    float inv = rowinv[row];
    __bf16* arow = attn + (size_t)b * NDIM * NDIM + (size_t)(n0 + row) * NDIM;
    for (int m = sub; m < NDIM; m += 16)
        arow[m] = f2bf(__expf(bf2f(S[row * NDIM + m]) - rm) * inv);
}

// ---------- Kernel 4: out[b,n,c] = (sum_m attn[b,n,m]*pvT[b,c,m]) / sqrt(N) ----------
// Block = (b, n-tile). The 16x2048 attn strip (64KB, contiguous) is async-staged into LDS
// once and shared by all 8 waves; wave w computes the 16(n) x 64(c) strip at c0 = w*64.
__global__ __launch_bounds__(256) void out_kernel(const __bf16* __restrict__ attn,
                                                  const __bf16* __restrict__ pvT,
                                                  float* __restrict__ out) {
    __shared__ __bf16 Atile[16 * NDIM];       // 64 KB
    const int lane = threadIdx.x & 31;
    const int wave = threadIdx.x >> 5;
    int nt = blockIdx.x & 127;
    int b  = blockIdx.x >> 7;
    int n0 = nt * 16, c0 = wave * 64;

    // Stage attn strip: 16 rows x 2048 cols bf16 == contiguous 64KB.
    {
        const char* Ag = (const char*)(attn + (size_t)b * NDIM * NDIM + (size_t)n0 * NDIM);
        unsigned l0 = lds_offset(Atile);
#pragma unroll
        for (int j = 0; j < 16; ++j) {
            int ch = threadIdx.x + j * 256;
            async_copy_b128(l0 + ch * 16, Ag + (size_t)ch * 16);
        }
        wait_asynccnt0();
    }
    __syncthreads();

    const __bf16* Bbase = pvT + (size_t)b * CDIM * NDIM + (size_t)c0 * NDIM;  // ld N

    v8f acc0 = {}, acc1 = {}, acc2 = {}, acc3 = {};
    for (int k = 0; k < NDIM; k += 32) {
        __builtin_prefetch(Bbase + k + 256, 0, 3);
        v16bf a  = load_frag_a(Atile + k, NDIM, lane);           // from LDS
        v16bf b0 = load_frag_b(Bbase + k,                     NDIM, lane);
        v16bf b1 = load_frag_b(Bbase + k + (size_t)16 * NDIM, NDIM, lane);
        v16bf b2 = load_frag_b(Bbase + k + (size_t)32 * NDIM, NDIM, lane);
        v16bf b3 = load_frag_b(Bbase + k + (size_t)48 * NDIM, NDIM, lane);
        acc0 = wmma_bf16(a, b0, acc0);
        acc1 = wmma_bf16(a, b1, acc1);
        acc2 = wmma_bf16(a, b2, acc2);
        acc3 = wmma_bf16(a, b3, acc3);
    }
    int nn = lane & 15;
    int mb = (lane >> 4) * 8;
    const float scale = 0.022097086912079608f;  // 1/sqrt(2048)
    v8f accs[4] = {acc0, acc1, acc2, acc3};
#pragma unroll
    for (int t = 0; t < 4; ++t) {
        float* o = out + (size_t)b * NDIM * CDIM + (size_t)n0 * CDIM + (c0 + t * 16 + nn);
#pragma unroll
        for (int r = 0; r < 8; ++r) o[(size_t)(mb + r) * CDIM] = accs[t][r] * scale;
    }
}

extern "C" void kernel_launch(void* const* d_in, const int* in_sizes, int n_in,
                              void* d_out, int out_size, void* d_ws, size_t ws_size,
                              hipStream_t stream) {
    (void)in_sizes; (void)n_in; (void)out_size; (void)ws_size;
    const float* q  = (const float*)d_in[0];
    const float* k  = (const float*)d_in[1];
    const float* v  = (const float*)d_in[2];
    const float* Wq = (const float*)d_in[3];
    const float* bq = (const float*)d_in[4];
    const float* Wk = (const float*)d_in[5];
    const float* bk = (const float*)d_in[6];
    const float* Wv = (const float*)d_in[7];
    const float* bv = (const float*)d_in[8];
    float* out = (float*)d_out;

    const size_t szX = (size_t)BDIM * CDIM * NDIM;   // 8,388,608
    const size_t szW = (size_t)NDIM * NDIM;          // 4,194,304
    const size_t szA = (size_t)BDIM * NDIM * NDIM;   // 33,554,432

    char* w = (char*)d_ws;
    __bf16* qh   = (__bf16*)w; w += szX * 2;
    __bf16* kh   = (__bf16*)w; w += szX * 2;
    __bf16* vh   = (__bf16*)w; w += szX * 2;
    __bf16* Wqh  = (__bf16*)w; w += szW * 2;
    __bf16* Wkh  = (__bf16*)w; w += szW * 2;
    __bf16* Wvh  = (__bf16*)w; w += szW * 2;
    __bf16* pq   = (__bf16*)w; w += szX * 2;   // [b][o][c]
    __bf16* pk   = (__bf16*)w; w += szX * 2;   // [b][o][c]
    __bf16* pvT  = (__bf16*)w; w += szX * 2;   // [b][c][o]
    __bf16* attn = (__bf16*)w; w += szA * 2;   // [b][n][m]

    // 1) convert to bf16
    {
        int gX = (int)((szX / 4 + 255) / 256);
        int gW = (int)((szW / 4 + 255) / 256);
        cvt_kernel<<<gX, 256, 0, stream>>>(q,  qh,  (int)szX);
        cvt_kernel<<<gX, 256, 0, stream>>>(k,  kh,  (int)szX);
        cvt_kernel<<<gX, 256, 0, stream>>>(v,  vh,  (int)szX);
        cvt_kernel<<<gW, 256, 0, stream>>>(Wq, Wqh, (int)szW);
        cvt_kernel<<<gW, 256, 0, stream>>>(Wk, Wkh, (int)szW);
        cvt_kernel<<<gW, 256, 0, stream>>>(Wv, Wvh, (int)szW);
    }

    // 2) projections: block = (b, o-tile) -> 8*128 = 1024 blocks, 8 waves split C
    proj_kernel<<<1024, 256, 0, stream>>>(Wqh, qh, bq, pq,  0);
    proj_kernel<<<1024, 256, 0, stream>>>(Wkh, kh, bk, pk,  0);
    proj_kernel<<<1024, 256, 0, stream>>>(Wvh, vh, bv, pvT, 1);

    // 3) energy + softmax: one block per (b, 16-row n tile) -> 1024 blocks
    energy_softmax_kernel<<<1024, 256, 0, stream>>>(pq, pk, attn);

    // 4) attn @ pv, scaled, fp32 out
    out_kernel<<<1024, 256, 0, stream>>>(attn, pvT, out);
}
